// CharNNClassifier_32040456028597
// MI455X (gfx1250) — matrix operand order repaired and checked by reading, not compiled, not run on hardware
//
#include <hip/hip_runtime.h>
#include <hip/hip_bf16.h>
#include <math.h>

// ---------------------------------------------------------------------------
// CharNN LSTM classifier for gfx1250 (MI455X), wave32 + WMMA bf16 + async LDS.
//
// Sizes: T=128, B=256, I=512, H=1024, V=128, O=64.
//
//   (a) G_emb[v][j] = emb[v]·w_ih[j] + b_ih[j] + b_hh[j]  (input GEMM folded
//       into a 128x4096 table since vocab is tiny)
//   (b) persistent kernel, 256 WGs x 128 thr, 128 steps, each step:
//       gates = G_emb[token] + h @ Whh^T via v_wmma_f32_16x16x32_bf16.
//       Whh B-tiles staged global->LDS with GLOBAL_LOAD_ASYNC_TO_LDS_B128
//       (ASYNCcnt), double-buffered; fully-unrolled K loop so accumulators
//       rename cleanly; branch-free LSTM cell; one device barrier per step.
//   (c) tiny fp32 head  out = h_last @ w_out^T + b_out.
// ---------------------------------------------------------------------------

#define T_STEPS 128
#define BATCH   256
#define IDIM    512
#define HDIM    1024
#define GDIM    (4 * HDIM)   // 4096
#define VOCAB   128
#define ODIM    64
#define NWG     256          // persistent workgroups
#define NTHR    128          // 4 wave32 per WG

typedef __attribute__((ext_vector_type(16))) __bf16 v16bf;
typedef __attribute__((ext_vector_type(8)))  float  v8f;

union V16 {            // 32 bytes: one WMMA 16-bit operand per lane
    uint4  u[2];
    v16bf  b;
};

// Branch-free activations: pure v_exp_f32 / v_rcp_f32, no lane divergence.
__device__ __forceinline__ float sigmoid_fast(float x) {
    return __fdividef(1.0f, 1.0f + __expf(-x));
}
__device__ __forceinline__ float tanh_fast(float x) {
    // 1 - 2/(1+e^{2x}): x->+inf => 1, x->-inf => -1; no branches.
    return 1.0f - __fdividef(2.0f, 1.0f + __expf(2.0f * x));
}

// CDNA5 async global->LDS copy, 16 bytes per lane, tracked by ASYNCcnt.
// volatile keeps issue order among the async ops and the asynccnt wait;
// no "memory" clobber: __syncthreads() provides the compiler-level fence
// for the LDS buffer handoff, so the scheduler stays free to pipeline.
__device__ __forceinline__ void async_copy_b128(unsigned dstLds, const void* src) {
    asm volatile("global_load_async_to_lds_b128 %0, %1, off"
                 :: "v"(dstLds), "v"(src));
}
__device__ __forceinline__ void wait_asynccnt0() {
    asm volatile("s_wait_asynccnt 0" ::: "memory");
}

// ---------------------------------------------------------------------------
// Prep 1: convert w_hh (4096x1024 f32, row-major [gateOut][k]) to bf16.
// ---------------------------------------------------------------------------
__global__ void k_convert_whh(const float* __restrict__ whh,
                              __bf16* __restrict__ whh_bf) {
    int i = blockIdx.x * blockDim.x + threadIdx.x;   // exactly 4H*H threads
    whh_bf[i] = (__bf16)whh[i];
}

// ---------------------------------------------------------------------------
// Prep 2: G_emb[v][j] = emb[v,:]·w_ih[j,:] + b_ih[j] + b_hh[j]
// ---------------------------------------------------------------------------
__global__ void k_embgates(const float* __restrict__ emb,
                           const float* __restrict__ w_ih,
                           const float* __restrict__ b_ih,
                           const float* __restrict__ b_hh,
                           float* __restrict__ gEmb) {
    int idx = blockIdx.x * blockDim.x + threadIdx.x;  // V*GDIM threads
    int v = idx >> 12;            // / 4096
    int j = idx & (GDIM - 1);     // % 4096
    const float4* e = (const float4*)(emb  + (size_t)v * IDIM);
    const float4* w = (const float4*)(w_ih + (size_t)j * IDIM);
    float acc = 0.0f;
#pragma unroll 4
    for (int i = 0; i < IDIM / 4; ++i) {
        float4 a = e[i], b = w[i];
        acc += a.x * b.x + a.y * b.y + a.z * b.z + a.w * b.w;
    }
    gEmb[idx] = acc + b_ih[j] + b_hh[j];
}

// ---------------------------------------------------------------------------
// Prep 3: initialize state: c <- c0, hbuf0 <- bf16(h0), hbuf1 <- 0, barrier<-0
// ---------------------------------------------------------------------------
__global__ void k_init_state(const float* __restrict__ h0,
                             const float* __restrict__ c0,
                             float* __restrict__ c_st,
                             __bf16* __restrict__ hbuf0,
                             __bf16* __restrict__ hbuf1,
                             unsigned* __restrict__ bar) {
    int i = blockIdx.x * blockDim.x + threadIdx.x;   // BATCH*HDIM threads
    c_st[i]  = c0[i];
    hbuf0[i] = (__bf16)h0[i];
    hbuf1[i] = (__bf16)0.0f;
    if (i == 0) { bar[0] = 0u; bar[1] = 0u; }
}

// ---------------------------------------------------------------------------
// Device-wide sense barrier (one per LSTM step).
// ---------------------------------------------------------------------------
__device__ __forceinline__ void grid_barrier(unsigned* bar, unsigned nwg) {
    __syncthreads();
    if (threadIdx.x == 0) {
        unsigned* cnt = bar;
        unsigned* gen = bar + 1;
        __threadfence();  // publish h/c stores at device scope
        unsigned g = __hip_atomic_load(gen, __ATOMIC_ACQUIRE, __HIP_MEMORY_SCOPE_AGENT);
        unsigned a = __hip_atomic_fetch_add(cnt, 1u, __ATOMIC_ACQ_REL, __HIP_MEMORY_SCOPE_AGENT);
        if (a == nwg - 1u) {
            __hip_atomic_store(cnt, 0u, __ATOMIC_RELAXED, __HIP_MEMORY_SCOPE_AGENT);
            __hip_atomic_fetch_add(gen, 1u, __ATOMIC_RELEASE, __HIP_MEMORY_SCOPE_AGENT);
        } else {
            while (__hip_atomic_load(gen, __ATOMIC_ACQUIRE, __HIP_MEMORY_SCOPE_AGENT) == g) {
                __builtin_amdgcn_s_sleep(1);
            }
        }
        __threadfence();
    }
    __syncthreads();
}

// ---------------------------------------------------------------------------
// Main persistent LSTM kernel.
//   grid = 256 WGs: wgM = blockIdx>>6 (4 batch quarters of 64 rows),
//                   wgN = blockIdx&63 (64 hidden slices of 16 cols).
//   block = 128 threads = 4 wave32; wave w owns batch rows
//   [batchBase+16w, batchBase+16w+16) x hidden cols [colBase, colBase+16)
//   and accumulates the 4 gate tiles (i,f,g,o) for that tile.
// ---------------------------------------------------------------------------
__global__ void __launch_bounds__(NTHR)
k_lstm(const int* __restrict__ tokens,          // [T, B]
       const float* __restrict__ gEmb,          // [V, 4H]
       const __bf16* __restrict__ whh_bf,       // [4H, H]
       float* __restrict__ c_st,                // [B, H]
       float* __restrict__ h_f32,               // [B, H] (written at t=T-1)
       __bf16* __restrict__ hbuf0,              // [B, H] ping
       __bf16* __restrict__ hbuf1,              // [B, H] pong
       unsigned* __restrict__ bar) {

    __shared__ uint4 ldsB[2][256];              // double-buffered, 2 x 4 KB

    const int tid       = threadIdx.x;
    const int lane      = tid & 31;
    const int waveId    = tid >> 5;             // 0..3
    const int wgN       = blockIdx.x & 63;
    const int wgM       = blockIdx.x >> 6;      // 0..3
    const int colBase   = wgN * 16;             // hidden column slice
    const int batchBase = wgM * 64;
    const int mBase     = batchBase + waveId * 16;

    // WMMA 16-bit A layout: lane L holds row M = L&15; its 16 elements are
    // K = kHalf..kHalf+7 and K = kHalf+16..kHalf+23, kHalf = (L<16 ? 0 : 8).
    const int aRow  = mBase + (lane & 15);
    const int aKoff = (lane >> 4) * 8;

    // Cooperative B staging: 256 x 16B per chunk (per-lane WMMA B layout),
    // 128 threads copy two each: slots q0 = tid (gates 0,1), q1 = tid+128.
    auto bSrc = [&](int q) -> const __bf16* {
        const int gate = q >> 6;
        const int r    = q & 63;
        const int ln   = r >> 1;
        const int half = r & 1;
        const int n    = ln & 15;
        const int koff = ((ln >> 4) << 3) + half * 16;
        return whh_bf + (size_t)(colBase + gate * HDIM + n) * HDIM + koff;
    };
    const __bf16* bPtr0 = bSrc(tid);
    const __bf16* bPtr1 = bSrc(tid + 128);
    // LDS byte offsets of this thread's two staging slots, per buffer.
    const unsigned ldsDst0[2] = { (unsigned)(size_t)&ldsB[0][tid],
                                  (unsigned)(size_t)&ldsB[1][tid] };
    const unsigned ldsDst1[2] = { (unsigned)(size_t)&ldsB[0][tid + 128],
                                  (unsigned)(size_t)&ldsB[1][tid + 128] };

    for (int t = 0; t < T_STEPS; ++t) {
        const __bf16* hRead  = (t & 1) ? hbuf1 : hbuf0;
        __bf16*       hWrite = (t & 1) ? hbuf0 : hbuf1;

        v8f acc[4] = {v8f{}, v8f{}, v8f{}, v8f{}};   // i, f, g, o tiles
        const __bf16* aPtr = hRead + (size_t)aRow * HDIM + aKoff;

        // Prologue: stage chunk 0 into buffer 0 (async, ASYNCcnt).
        async_copy_b128(ldsDst0[0], bPtr0);
        async_copy_b128(ldsDst1[0], bPtr1);

#pragma unroll
        for (int kc = 0; kc < HDIM / 32; ++kc) {     // 32 chunks of K=32
            const int k0  = kc * 32;
            const int cur = kc & 1;
            wait_asynccnt0();  // buf[cur] copies landed in LDS
            __syncthreads();   // all waves staged; prev reads of buf[cur^1] done

            if (kc + 1 < HDIM / 32) {                // stage next chunk async
                const int k1 = k0 + 32;
                async_copy_b128(ldsDst0[cur ^ 1], bPtr0 + k1);
                async_copy_b128(ldsDst1[cur ^ 1], bPtr1 + k1);
            }

            V16 a;
            a.u[0] = *(const uint4*)(aPtr + k0);
            a.u[1] = *(const uint4*)(aPtr + k0 + 16);

#pragma unroll
            for (int g = 0; g < 4; ++g) {
                V16 b;
                b.u[0] = ldsB[cur][g * 64 + lane * 2];
                b.u[1] = ldsB[cur][g * 64 + lane * 2 + 1];
                acc[g] = __builtin_amdgcn_wmma_f32_16x16x32_bf16(
                    false, a.b, false, b.b, (short)0, acc[g], false, false);
            }
        }

        // Elementwise LSTM cell. C/D layout: VGPR r, lane L ->
        //   row = r + (L>=16 ? 8 : 0), col = L & 15. Identical for all gates.
        const int  col    = colBase + (lane & 15);
        const int  rowOff = (lane >> 4) * 8;
        const int* tok    = tokens + (size_t)t * BATCH;

#pragma unroll
        for (int r = 0; r < 8; ++r) {
            const int row = mBase + rowOff + r;       // batch index
            const int v   = tok[row];
            const float* ge = gEmb + (size_t)v * GDIM;

            float xi = acc[0][r] + ge[col];
            float xf = acc[1][r] + ge[col + HDIM];
            float xg = acc[2][r] + ge[col + 2 * HDIM];
            float xo = acc[3][r] + ge[col + 3 * HDIM];

            float ii = sigmoid_fast(xi);
            float ff = sigmoid_fast(xf);
            float gg = tanh_fast(xg);
            float oo = sigmoid_fast(xo);

            const size_t idx = (size_t)row * HDIM + col;
            float cc = ff * c_st[idx] + ii * gg;
            c_st[idx] = cc;
            float hh = oo * tanh_fast(cc);
            hWrite[idx] = (__bf16)hh;
            if (t == T_STEPS - 1) h_f32[idx] = hh;
        }

        grid_barrier(bar, NWG);
    }
}

// ---------------------------------------------------------------------------
// Head: out[b][o] = h_last[b,:]·w_out[o,:] + b_out[o]   (256x64, K=1024)
// ---------------------------------------------------------------------------
__global__ void k_head(const float* __restrict__ h_f32,
                       const float* __restrict__ w_out,
                       const float* __restrict__ b_out,
                       float* __restrict__ out) {
    const int b = blockIdx.x;       // 256 blocks
    const int o = threadIdx.x;      // 64 threads
    const float4* hv = (const float4*)(h_f32 + (size_t)b * HDIM);
    const float4* wv = (const float4*)(w_out + (size_t)o * HDIM);
    float acc = 0.0f;
#pragma unroll 4
    for (int i = 0; i < HDIM / 4; ++i) {
        float4 a = hv[i], w = wv[i];
        acc += a.x * w.x + a.y * w.y + a.z * w.z + a.w * w.w;
    }
    out[b * ODIM + o] = acc + b_out[o];
}

// ---------------------------------------------------------------------------
// Launch
// ---------------------------------------------------------------------------
extern "C" void kernel_launch(void* const* d_in, const int* in_sizes, int n_in,
                              void* d_out, int out_size, void* d_ws, size_t ws_size,
                              hipStream_t stream) {
    const int*   input = (const int*)  d_in[0];   // [T, B]
    const float* h0    = (const float*)d_in[1];   // [1, B, H]
    const float* c0    = (const float*)d_in[2];   // [1, B, H]
    const float* emb   = (const float*)d_in[3];   // [V, I]
    const float* w_ih  = (const float*)d_in[4];   // [4H, I]
    const float* w_hh  = (const float*)d_in[5];   // [4H, H]
    const float* b_ih  = (const float*)d_in[6];   // [4H]
    const float* b_hh  = (const float*)d_in[7];   // [4H]
    const float* w_out = (const float*)d_in[8];   // [O, H]
    const float* b_out = (const float*)d_in[9];   // [O]
    float* out = (float*)d_out;                   // [B, O]

    // Workspace carve-up (all offsets 256B-aligned).
    char* ws = (char*)d_ws;
    size_t off = 0;
    __bf16*   whh_bf = (__bf16*)(ws + off);  off += (size_t)GDIM * HDIM * 2;   // 8 MB
    float*    gEmb   = (float*) (ws + off);  off += (size_t)VOCAB * GDIM * 4;  // 2 MB
    float*    c_st   = (float*) (ws + off);  off += (size_t)BATCH * HDIM * 4;  // 1 MB
    float*    h_f32  = (float*) (ws + off);  off += (size_t)BATCH * HDIM * 4;  // 1 MB
    __bf16*   hbuf0  = (__bf16*)(ws + off);  off += (size_t)BATCH * HDIM * 2;  // 0.5 MB
    __bf16*   hbuf1  = (__bf16*)(ws + off);  off += (size_t)BATCH * HDIM * 2;  // 0.5 MB
    unsigned* bar    = (unsigned*)(ws + off); off += 256;

    // Prep
    k_convert_whh<<<(GDIM * HDIM) / 256, 256, 0, stream>>>(w_hh, whh_bf);
    k_embgates<<<(VOCAB * GDIM) / 256, 256, 0, stream>>>(emb, w_ih, b_ih, b_hh, gEmb);
    k_init_state<<<(BATCH * HDIM) / 256, 256, 0, stream>>>(h0, c0, c_st, hbuf0, hbuf1, bar);

    // Persistent recurrent kernel: 256 WGs (4 batch quarters x 64 hidden slices)
    k_lstm<<<NWG, NTHR, 0, stream>>>(input, gEmb, whh_bf, c_st, h_f32,
                                     hbuf0, hbuf1, bar);

    // Head
    k_head<<<BATCH, ODIM, 0, stream>>>(h_f32, w_out, b_out, out);
}
